// TreeLSTM_14491219657207
// MI455X (gfx1250) — matrix operand order repaired and checked by reading, not compiled, not run on hardware
//
#include <hip/hip_runtime.h>
#include <hip/hip_bf16.h>

#define NNODES 100000
#define NEDGES 3200000
#define NIN    128
#define NHID   128
#define NOUT   64
#define NGATE  512   // 4*NHID
#define HOPS   10

typedef __attribute__((ext_vector_type(16))) __bf16 bf16x16;
typedef __attribute__((ext_vector_type(8)))  float  f32x8;

__device__ __forceinline__ f32x8 wmma_bf16(bf16x16 a, bf16x16 b, f32x8 c) {
  return __builtin_amdgcn_wmma_f32_16x16x32_bf16(false, a, false, b, (short)0, c,
                                                 false, false);
}

// A-matrix 16x32 bf16 tile, row-major src [M, ld].
// Lane<16: row m0+lane, K = k0..k0+7 (VGPR0-3) and k0+16..k0+23 (VGPR4-7).
// Lane>=16: row m0+lane-16, K = k0+8..k0+15 and k0+24..k0+31.
__device__ __forceinline__ bf16x16 load_a_tile(const __hip_bfloat16* __restrict__ src,
                                               int ld, int m0, int k0, int lane) {
  int row   = m0 + (lane & 15);
  int kbase = k0 + ((lane >> 4) << 3);
  const __hip_bfloat16* p = src + (size_t)row * ld + kbase;
  union { bf16x16 v; uint4 q[2]; } u;
  u.q[0] = *(const uint4*)(p);
  u.q[1] = *(const uint4*)(p + 16);
  return u.v;
}

// B-matrix 32x16 bf16 tile with B = W^T, W row-major [N, ld] (ld = K dim).
// Lane<16: col n0+lane, K = k0..k0+7 and k0+16..k0+23.
// Lane>=16: col n0+lane-16, K = k0+8..k0+15 and k0+24..k0+31.
__device__ __forceinline__ bf16x16 load_b_tile(const __hip_bfloat16* __restrict__ W,
                                               int ld, int n0, int k0, int lane) {
  int col   = n0 + (lane & 15);
  int kbase = k0 + ((lane >> 4) << 3);
  const __hip_bfloat16* p = W + (size_t)col * ld + kbase;
  union { bf16x16 v; uint4 q[2]; } u;
  u.q[0] = *(const uint4*)(p);
  u.q[1] = *(const uint4*)(p + 16);
  return u.v;
}

__device__ __forceinline__ float sigmoidf_fast(float x) {
  return 1.0f / (1.0f + __expf(-x));
}

// ---------------------------------------------------------------- converts
__global__ __launch_bounds__(256) void cvt_f32_bf16_kernel(
    const float* __restrict__ src, __hip_bfloat16* __restrict__ dst, int n) {
  int i = blockIdx.x * blockDim.x + threadIdx.x;
  if (i < n) dst[i] = __float2bfloat16(src[i]);
}

// ---------------------------------------------------------------- gx = x @ W_ih^T + b_ih + b_hh
// 512 threads = 16 waves. Wave wv owns cols [32wv, 32wv+32) and keeps its
// W columns (2 N-tiles x K=128 -> 64 VGPRs) resident for the whole kernel.
// Persistent grid-stride over 32-row blocks. No LDS needed.
__global__ __launch_bounds__(512, 1) void gemm_gx_kernel(
    const __hip_bfloat16* __restrict__ A,    // x bf16 [N,128]
    const __hip_bfloat16* __restrict__ W,    // W_ih bf16 [512,128]
    const float* __restrict__ b_ih,
    const float* __restrict__ b_hh,
    float* __restrict__ gx) {                // [N,512]
  const int lane = threadIdx.x & 31;
  const int wv   = threadIdx.x >> 5;
  const int ccol = lane & 15;
  const int crow = (lane >> 4) << 3;
  const int n0   = wv * 32;

  bf16x16 Breg[4][2];
#pragma unroll
  for (int ks = 0; ks < 4; ks++)
#pragma unroll
    for (int nt = 0; nt < 2; nt++)
      Breg[ks][nt] = load_b_tile(W, NIN, n0 + nt * 16, ks * 32, lane);

  float bias[2];
#pragma unroll
  for (int nt = 0; nt < 2; nt++) {
    int col = n0 + nt * 16 + ccol;
    bias[nt] = b_ih[col] + b_hh[col];
  }

  for (int rb = blockIdx.x; rb < NNODES / 32; rb += gridDim.x) {
    const int m0 = rb * 32;
    __builtin_prefetch(A + (size_t)(m0 + 32 * (int)gridDim.x < NNODES
                                        ? m0 + 32 * (int)gridDim.x : m0) * NIN, 0, 0);
    f32x8 acc[2][2] = {};
#pragma unroll
    for (int ks = 0; ks < 4; ks++) {
      bf16x16 a0 = load_a_tile(A, NIN, m0, ks * 32, lane);
      bf16x16 a1 = load_a_tile(A, NIN, m0 + 16, ks * 32, lane);
#pragma unroll
      for (int nt = 0; nt < 2; nt++) {
        acc[0][nt] = wmma_bf16(a0, Breg[ks][nt], acc[0][nt]);
        acc[1][nt] = wmma_bf16(a1, Breg[ks][nt], acc[1][nt]);
      }
    }
#pragma unroll
    for (int mt = 0; mt < 2; mt++)
#pragma unroll
      for (int nt = 0; nt < 2; nt++) {
        int col = n0 + nt * 16 + ccol;
#pragma unroll
        for (int r = 0; r < 8; r++) {
          int row = m0 + mt * 16 + crow + r;
          gx[(size_t)row * NGATE + col] = acc[mt][nt][r] + bias[nt];
        }
      }
  }
}

// ---------------------------------------------------------------- gates = h@W_hh^T, fused LSTM
// Same register-resident-B scheme; LDS only holds the 32x512 f32 gate stage
// (64KB) for the cross-wave i/f/g/o shuffle.
__global__ __launch_bounds__(512, 1) void gate_lstm_kernel(
    const __hip_bfloat16* __restrict__ Hbf,  // [N,128]
    const __hip_bfloat16* __restrict__ W,    // W_hh bf16 [512,128]
    const float* __restrict__ gx,            // [N,512]
    float* __restrict__ c,                   // [N,128] in/out
    float* __restrict__ h_out) {             // [N,128]
  extern __shared__ char smem[];
  float* glds = (float*)smem;                // 32*512 floats

  const int lane = threadIdx.x & 31;
  const int wv   = threadIdx.x >> 5;
  const int ccol = lane & 15;
  const int crow = (lane >> 4) << 3;
  const int n0   = wv * 32;

  bf16x16 Breg[4][2];
#pragma unroll
  for (int ks = 0; ks < 4; ks++)
#pragma unroll
    for (int nt = 0; nt < 2; nt++)
      Breg[ks][nt] = load_b_tile(W, NHID, n0 + nt * 16, ks * 32, lane);

  for (int rb = blockIdx.x; rb < NNODES / 32; rb += gridDim.x) {
    const int m0 = rb * 32;
    __builtin_prefetch(Hbf + (size_t)(m0 + 32 * (int)gridDim.x < NNODES
                                          ? m0 + 32 * (int)gridDim.x : m0) * NHID, 0, 0);
    f32x8 acc[2][2] = {};
#pragma unroll
    for (int ks = 0; ks < 4; ks++) {
      bf16x16 a0 = load_a_tile(Hbf, NHID, m0, ks * 32, lane);
      bf16x16 a1 = load_a_tile(Hbf, NHID, m0 + 16, ks * 32, lane);
#pragma unroll
      for (int nt = 0; nt < 2; nt++) {
        acc[0][nt] = wmma_bf16(a0, Breg[ks][nt], acc[0][nt]);
        acc[1][nt] = wmma_bf16(a1, Breg[ks][nt], acc[1][nt]);
      }
    }
#pragma unroll
    for (int mt = 0; mt < 2; mt++)
#pragma unroll
      for (int nt = 0; nt < 2; nt++) {
        int col = n0 + nt * 16 + ccol;
#pragma unroll
        for (int r = 0; r < 8; r++)
          glds[(mt * 16 + crow + r) * NGATE + col] = acc[mt][nt][r];
      }
    __syncthreads();

    // gating: 32 nodes x 128 hidden = 4096 elems over 512 threads
    for (int idx = threadIdx.x; idx < 32 * NHID; idx += blockDim.x) {
      int r = idx >> 7;
      int j = idx & (NHID - 1);
      int node = m0 + r;
      const float* gxr = gx + (size_t)node * NGATE;
      float gi = glds[r * NGATE + j]            + gxr[j];
      float gf = glds[r * NGATE + NHID + j]     + gxr[NHID + j];
      float gg = glds[r * NGATE + 2 * NHID + j] + gxr[2 * NHID + j];
      float go = glds[r * NGATE + 3 * NHID + j] + gxr[3 * NHID + j];
      float iv = sigmoidf_fast(gi);
      float fv = sigmoidf_fast(gf);
      float gv = tanhf(gg);
      float ov = sigmoidf_fast(go);
      size_t o = (size_t)node * NHID + j;
      float cv = fv * c[o] + iv * gv;
      c[o] = cv;
      h_out[o] = ov * tanhf(cv);
    }
    __syncthreads();
  }
}

// ---------------------------------------------------------------- SpMM: hs[row] += val * h[col]
// one wave per edge, 4 floats per lane (32*4 = 128 features)
__global__ __launch_bounds__(256) void spmm_kernel(
    const int* __restrict__ erow, const int* __restrict__ ecol,
    const float* __restrict__ eval,
    const float* __restrict__ h, float* __restrict__ hs, int E) {
  int t = blockIdx.x * blockDim.x + threadIdx.x;
  int e = t >> 5;
  if (e >= E) return;
  int lane = t & 31;
  int rdst = erow[e];
  int csrc = ecol[e];
  float v = eval[e];
  float4 hv = *(const float4*)(h + (size_t)csrc * NHID + lane * 4);
  float* dst = hs + (size_t)rdst * NHID + lane * 4;
  atomicAdd(dst + 0, v * hv.x);
  atomicAdd(dst + 1, v * hv.y);
  atomicAdd(dst + 2, v * hv.z);
  atomicAdd(dst + 3, v * hv.w);
}

// ---------------------------------------------------------------- out = relu(h) @ fc_w^T + fc_b
// 256 threads = 8 waves; 2 waves per M-tile, each wave owns 2 N-tiles (N=64).
__global__ __launch_bounds__(256) void final_fc_kernel(
    const float* __restrict__ H,              // [N,128]
    const __hip_bfloat16* __restrict__ Wfc,   // fc_w bf16 [64,128]
    const float* __restrict__ fcb,
    float* __restrict__ out) {                // [N,64]
  const int lane = threadIdx.x & 31;
  const int wv   = threadIdx.x >> 5;
  const int mt   = blockIdx.x * 4 + (wv >> 1);
  const int n0   = (wv & 1) * 32;
  if (mt >= NNODES / 16) return;
  const int m0   = mt * 16;
  const int ccol = lane & 15;
  const int crow = (lane >> 4) << 3;

  f32x8 acc[2] = {};
#pragma unroll
  for (int ks = 0; ks < 4; ks++) {
    const int k = ks * 32;
    // A tile: relu + cvt from f32 on the fly
    int row = m0 + (lane & 15);
    int kb  = k + ((lane >> 4) << 3);
    const float* p = H + (size_t)row * NHID + kb;
    union { bf16x16 v; __hip_bfloat16 e[16]; } u;
#pragma unroll
    for (int i = 0; i < 8; i++) {
      float v0 = p[i];
      u.e[i] = __float2bfloat16(v0 > 0.f ? v0 : 0.f);
    }
#pragma unroll
    for (int i = 0; i < 8; i++) {
      float v0 = p[16 + i];
      u.e[8 + i] = __float2bfloat16(v0 > 0.f ? v0 : 0.f);
    }
#pragma unroll
    for (int nt = 0; nt < 2; nt++) {
      bf16x16 b = load_b_tile(Wfc, NHID, n0 + nt * 16, k, lane);
      acc[nt] = wmma_bf16(u.v, b, acc[nt]);
    }
  }
#pragma unroll
  for (int nt = 0; nt < 2; nt++) {
    int col = n0 + nt * 16 + ccol;
    float bias = fcb[col];
#pragma unroll
    for (int r = 0; r < 8; r++)
      out[(size_t)(m0 + crow + r) * NOUT + col] = acc[nt][r] + bias;
  }
}

// ----------------------------------------------------------------
extern "C" void kernel_launch(void* const* d_in, const int* in_sizes, int n_in,
                              void* d_out, int out_size, void* d_ws, size_t ws_size,
                              hipStream_t stream) {
  (void)in_sizes; (void)n_in; (void)out_size; (void)ws_size;
  const float* x     = (const float*)d_in[0];
  const int*   erow  = (const int*)d_in[1];
  const int*   ecol  = (const int*)d_in[2];
  const float* eval  = (const float*)d_in[3];
  const float* W_ih  = (const float*)d_in[4];
  const float* W_hh  = (const float*)d_in[5];
  const float* b_ih  = (const float*)d_in[6];
  const float* b_hh  = (const float*)d_in[7];
  const float* fc_w  = (const float*)d_in[8];
  const float* fc_b  = (const float*)d_in[9];
  float* out = (float*)d_out;

  // workspace carve-out (256B aligned)
  char* ws = (char*)d_ws;
  size_t off = 0;
  auto carve = [&](size_t bytes) -> void* {
    void* p = ws + off;
    off += (bytes + 255) & ~(size_t)255;
    return p;
  };
  __hip_bfloat16* x_bf   = (__hip_bfloat16*)carve((size_t)NNODES * NIN * 2);
  __hip_bfloat16* Wih_bf = (__hip_bfloat16*)carve((size_t)NGATE * NIN * 2);
  __hip_bfloat16* Whh_bf = (__hip_bfloat16*)carve((size_t)NGATE * NHID * 2);
  __hip_bfloat16* Wfc_bf = (__hip_bfloat16*)carve((size_t)NOUT * NHID * 2);
  __hip_bfloat16* h_bf   = (__hip_bfloat16*)carve((size_t)NNODES * NHID * 2);
  float* gx     = (float*)carve((size_t)NNODES * NGATE * 4);
  float* c_st   = (float*)carve((size_t)NNODES * NHID * 4);
  float* h_gate = (float*)carve((size_t)NNODES * NHID * 4);
  float* hs     = (float*)carve((size_t)NNODES * NHID * 4);

  // precision conversions
  cvt_f32_bf16_kernel<<<(NNODES * NIN + 255) / 256, 256, 0, stream>>>(x, x_bf, NNODES * NIN);
  cvt_f32_bf16_kernel<<<(NGATE * NIN + 255) / 256, 256, 0, stream>>>(W_ih, Wih_bf, NGATE * NIN);
  cvt_f32_bf16_kernel<<<(NGATE * NHID + 255) / 256, 256, 0, stream>>>(W_hh, Whh_bf, NGATE * NHID);
  cvt_f32_bf16_kernel<<<(NOUT * NHID + 255) / 256, 256, 0, stream>>>(fc_w, Wfc_bf, NOUT * NHID);

  hipMemsetAsync(c_st, 0, (size_t)NNODES * NHID * 4, stream);
  hipMemsetAsync(h_bf, 0, (size_t)NNODES * NHID * 2, stream);

  // hop-invariant gx = x @ W_ih^T + b_ih + b_hh
  gemm_gx_kernel<<<1024, 512, 0, stream>>>(x_bf, Wih_bf, b_ih, b_hh, gx);

  const size_t gate_lds = (size_t)32 * NGATE * 4;  // 64KB gate stage
  for (int hop = 0; hop < HOPS; hop++) {
    gate_lstm_kernel<<<1024, 512, gate_lds, stream>>>(h_bf, Whh_bf, gx, c_st, h_gate);
    hipMemsetAsync(hs, 0, (size_t)NNODES * NHID * 4, stream);
    spmm_kernel<<<(NEDGES * 32 + 255) / 256, 256, 0, stream>>>(erow, ecol, eval,
                                                               h_gate, hs, NEDGES);
    cvt_f32_bf16_kernel<<<(NNODES * NHID + 255) / 256, 256, 0, stream>>>(hs, h_bf,
                                                                         NNODES * NHID);
  }

  final_fc_kernel<<<(NNODES / 16 + 3) / 4, 256, 0, stream>>>(hs, Wfc_bf, fc_b, out);
}
